// GCN_16716012716713
// MI455X (gfx1250) — matrix operand order repaired and checked by reading, not compiled, not run on hardware
//
#include <hip/hip_runtime.h>
#include <hip/hip_bf16.h>

typedef __attribute__((ext_vector_type(16))) __bf16 v16bf;
typedef __attribute__((ext_vector_type(8)))  __bf16 v8bf;
typedef __attribute__((ext_vector_type(4)))  __bf16 v4bf;
typedef __attribute__((ext_vector_type(8)))  float  v8f;

#define N_NODES 100000
#define N_EDGES 1600000
#define F_IN    512
#define F_HID   128
#define F_OUT   64

// ---------------------------------------------------------------------------
// Degree / normalization kernels
// ---------------------------------------------------------------------------
__global__ __launch_bounds__(256) void init_deg_kernel(float* deg, int n) {
    int i = blockIdx.x * blockDim.x + threadIdx.x;
    if (i < n) deg[i] = 1.0f;  // self-loop contributes 1
}

__global__ __launch_bounds__(256) void count_deg_kernel(const int* __restrict__ dst,
                                                        float* __restrict__ deg, int e) {
    int i = blockIdx.x * blockDim.x + threadIdx.x;
    if (i < e) atomicAdd(&deg[dst[i]], 1.0f);
}

__global__ __launch_bounds__(256) void rsqrt_kernel(float* deg, int n) {
    int i = blockIdx.x * blockDim.x + threadIdx.x;
    if (i < n) deg[i] = rsqrtf(deg[i]);  // deg >= 1 always
}

// ---------------------------------------------------------------------------
// One-off: Wt[n*K + k] = bf16(W[k*N + n])   (transpose + downconvert weights)
// ---------------------------------------------------------------------------
__global__ __launch_bounds__(256)
void convert_wT_kernel(const float* __restrict__ W, __bf16* __restrict__ Wt,
                       int K, int N) {
    int i = blockIdx.x * blockDim.x + threadIdx.x;
    if (i >= K * N) return;
    int k = i / N, n = i % N;
    Wt[(size_t)n * K + k] = (__bf16)W[i];
}

// ---------------------------------------------------------------------------
// GEMM: C[M x BN] = A[M x K] * W[K x BN], fp32 A/C, bf16 WMMA compute.
// Wt is W transposed (BN x K) in bf16.
// Block: 256 threads = 8 waves. Tile: 64 rows x BN cols, BK = 32.
// 2 waves per 16-row M-tile; each wave owns BN/32 16-col N-tiles.
// LDS row stride = 40 bf16 (80 B): 16B-aligned chunks, conflict-free fan-out.
// ---------------------------------------------------------------------------
template <int BN>
__global__ __launch_bounds__(256)
void gemm_bf16_wmma(const float* __restrict__ A, const __bf16* __restrict__ Wt,
                    float* __restrict__ C, int M, int K) {
    constexpr int NTPW = BN / 32;   // N-tiles per wave
    constexpr int STR  = 40;        // LDS row stride in bf16 elements
    __shared__ __align__(16) __bf16 As[64 * STR];  // 64 rows x 32 K
    __shared__ __align__(16) __bf16 Bs[BN * STR];  // BN rows x 32 K (K-major!)

    const int tid   = threadIdx.x;
    const int lane  = tid & 31;
    const int wv    = tid >> 5;
    const int mtile = wv >> 1;   // 0..3
    const int half  = wv & 1;    // which N-half of the tile
    const int rowBase = blockIdx.x * 64;

    v8f acc[NTPW];
#pragma unroll
    for (int t = 0; t < NTPW; ++t) acc[t] = (v8f){0, 0, 0, 0, 0, 0, 0, 0};

    for (int k0 = 0; k0 < K; k0 += 32) {
        // --- Stage A tile (64 rows x 32 K) fp32 -> bf16, float4-vectorized
#pragma unroll
        for (int j = 0; j < 2; ++j) {
            int idx = tid + 256 * j;            // 512 float4 loads
            int r = idx >> 3, c4 = (idx & 7) << 2;
            int row = rowBase + r;
            float4 v = (row < M)
                ? *(const float4*)(A + (size_t)row * K + (k0 + c4))
                : make_float4(0.f, 0.f, 0.f, 0.f);
            *(v4bf*)(As + r * STR + c4) =
                (v4bf){(__bf16)v.x, (__bf16)v.y, (__bf16)v.z, (__bf16)v.w};
        }
        // --- Stage B tile (BN rows x 32 K) bf16 16B copies from Wt
#pragma unroll
        for (int j = 0; j < BN / 64; ++j) {
            int idx = tid + 256 * j;            // BN*4 chunks of 8 bf16
            int n = idx >> 2, ch = (idx & 3) << 3;
            *(v8bf*)(Bs + n * STR + ch) =
                *(const v8bf*)(Wt + (size_t)n * K + k0 + ch);
        }
        __syncthreads();

        // --- A fragment (16x32): lane L: M=L%16; elems 0-7 -> K=kg..kg+7,
        //     elems 8-15 -> K=kg+16..kg+23, kg=(L/16)*8.  Two b128 LDS loads.
        v16bf a;
        {
            int m  = mtile * 16 + (lane & 15);
            int ka = lane >> 4;                         // chunk 0 or 1
            const v8bf* pa = (const v8bf*)(As + m * STR);
            v8bf alo = pa[ka];
            v8bf ahi = pa[2 + ka];
            a = __builtin_shufflevector(alo, ahi, 0, 1, 2, 3, 4, 5, 6, 7,
                                        8, 9, 10, 11, 12, 13, 14, 15);
        }

#pragma unroll
        for (int t = 0; t < NTPW; ++t) {
            // --- B fragment (32x16): lane L: N=L%16; elems 0-7 -> K=kg..kg+7,
            //     elems 8-15 -> K=kg+16..kg+23, kg=(L/16)*8. Two b128 LDS loads.
            int nBase = (half * NTPW + t) * 16;
            int n  = nBase + (lane & 15);
            int kb = lane >> 4;                         // chunk 0 or 1
            const v8bf* pb = (const v8bf*)(Bs + n * STR);
            v8bf blo = pb[kb];
            v8bf bhi = pb[kb + 2];
            v16bf b = __builtin_shufflevector(blo, bhi, 0, 1, 2, 3, 4, 5, 6, 7,
                                              8, 9, 10, 11, 12, 13, 14, 15);

            acc[t] = __builtin_amdgcn_wmma_f32_16x16x32_bf16(
                false, a, false, b, (short)0, acc[t], false, false);
        }
        __syncthreads();
    }

    // --- Writeback: C/D layout — VGPR r: M = (lane/16)*8 + r, N = lane%16
#pragma unroll
    for (int t = 0; t < NTPW; ++t) {
        int col = (half * NTPW + t) * 16 + (lane & 15);
#pragma unroll
        for (int r = 0; r < 8; ++r) {
            int row = rowBase + mtile * 16 + ((lane >> 4) << 3) + r;
            if (row < M) C[(size_t)row * BN + col] = acc[t][r];
        }
    }
}

// ---------------------------------------------------------------------------
// Edge scatter: one wave per edge; agg[dst] += h[src] * dinv[src]*dinv[dst]
// h and agg stay resident in the 192MB L2 (51MB / 26MB working sets).
// ---------------------------------------------------------------------------
template <int F>
__global__ __launch_bounds__(256)
void scatter_edges(const int* __restrict__ src, const int* __restrict__ dst,
                   const float* __restrict__ dinv, const float* __restrict__ h,
                   float* __restrict__ agg, int E) {
    int edge = (blockIdx.x * blockDim.x + threadIdx.x) >> 5;
    int lane = threadIdx.x & 31;
    if (edge >= E) return;
    int s = src[edge], d = dst[edge];
    float norm = dinv[s] * dinv[d];
    const float* hs = h + (size_t)s * F;
    float* ad = agg + (size_t)d * F;
#pragma unroll
    for (int f = lane; f < F; f += 32)
        atomicAdd(&ad[f], hs[f] * norm);
}

// ---------------------------------------------------------------------------
// Self-loop + bias (+ optional ReLU): out = [relu](agg + h*dinv^2 + b)
// Safe for out == agg (element-wise in place).
// ---------------------------------------------------------------------------
template <int F, bool RELU>
__global__ __launch_bounds__(256)
void finish_kernel(const float* __restrict__ agg, const float* __restrict__ h,
                   const float* __restrict__ dinv, const float* __restrict__ b,
                   float* __restrict__ out, int n) {
    int i = blockIdx.x * blockDim.x + threadIdx.x;
    if (i >= n * F) return;
    int node = i / F, f = i % F;
    float s = dinv[node];
    float v = agg[i] + h[i] * s * s + b[f];
    if (RELU) v = fmaxf(v, 0.0f);
    out[i] = v;
}

// ---------------------------------------------------------------------------
extern "C" void kernel_launch(void* const* d_in, const int* in_sizes, int n_in,
                              void* d_out, int out_size, void* d_ws, size_t ws_size,
                              hipStream_t stream) {
    (void)in_sizes; (void)n_in; (void)out_size; (void)ws_size;

    const float* x  = (const float*)d_in[0];
    const int*   ei = (const int*)d_in[1];
    const float* W1 = (const float*)d_in[2];
    const float* b1 = (const float*)d_in[3];
    const float* W2 = (const float*)d_in[4];
    const float* b2 = (const float*)d_in[5];
    float* out = (float*)d_out;

    const int* srcI = ei;             // edge_index[0]
    const int* dstI = ei + N_EDGES;   // edge_index[1]

    char* ws = (char*)d_ws;
    float*  deg  = (float*)ws;  ws += sizeof(float) * (size_t)N_NODES;          // -> dinv
    float*  h1   = (float*)ws;  ws += sizeof(float) * (size_t)N_NODES * F_HID;
    float*  agg1 = (float*)ws;  ws += sizeof(float) * (size_t)N_NODES * F_HID;  // -> relu out
    float*  h2   = (float*)ws;  ws += sizeof(float) * (size_t)N_NODES * F_OUT;
    float*  agg2 = (float*)ws;  ws += sizeof(float) * (size_t)N_NODES * F_OUT;
    __bf16* Wt1  = (__bf16*)ws; ws += sizeof(__bf16) * (size_t)F_IN * F_HID;
    __bf16* Wt2  = (__bf16*)ws; ws += sizeof(__bf16) * (size_t)F_HID * F_OUT;

    hipMemsetAsync(agg1, 0, sizeof(float) * (size_t)N_NODES * F_HID, stream);
    hipMemsetAsync(agg2, 0, sizeof(float) * (size_t)N_NODES * F_OUT, stream);

    // Weight transpose + bf16 downconvert (tiny, one-off)
    convert_wT_kernel<<<(F_IN * F_HID + 255) / 256, 256, 0, stream>>>(W1, Wt1, F_IN, F_HID);
    convert_wT_kernel<<<(F_HID * F_OUT + 255) / 256, 256, 0, stream>>>(W2, Wt2, F_HID, F_OUT);

    // Normalization
    init_deg_kernel<<<(N_NODES + 255) / 256, 256, 0, stream>>>(deg, N_NODES);
    count_deg_kernel<<<(N_EDGES + 255) / 256, 256, 0, stream>>>(dstI, deg, N_EDGES);
    rsqrt_kernel<<<(N_NODES + 255) / 256, 256, 0, stream>>>(deg, N_NODES);

    // Layer 1: h1 = x @ W1 ; aggregate ; relu(+ self-loop + bias) in place -> agg1
    gemm_bf16_wmma<F_HID><<<(N_NODES + 63) / 64, 256, 0, stream>>>(x, Wt1, h1, N_NODES, F_IN);
    scatter_edges<F_HID><<<(N_EDGES * 32 + 255) / 256, 256, 0, stream>>>(srcI, dstI, deg, h1, agg1, N_EDGES);
    finish_kernel<F_HID, true><<<((N_NODES * F_HID) + 255) / 256, 256, 0, stream>>>(
        agg1, h1, deg, b1, agg1, N_NODES);

    // Layer 2: h2 = agg1 @ W2 ; aggregate ; + self-loop + bias -> out
    gemm_bf16_wmma<F_OUT><<<(N_NODES + 63) / 64, 256, 0, stream>>>(agg1, Wt2, h2, N_NODES, F_HID);
    scatter_edges<F_OUT><<<(N_EDGES * 32 + 255) / 256, 256, 0, stream>>>(srcI, dstI, deg, h2, agg2, N_EDGES);
    finish_kernel<F_OUT, false><<<((N_NODES * F_OUT) + 255) / 256, 256, 0, stream>>>(
        agg2, h2, deg, b2, out, N_NODES);
}
